// Kmeans_38113539784874
// MI455X (gfx1250) — matrix operand order repaired and checked by reading, not compiled, not run on hardware
//
#include <hip/hip_runtime.h>

// ---------------------------------------------------------------------------
// K-means assignment for MI455X (gfx1250, wave32, WMMA).
//   argmin_k ||x_n - c_k||^2 == argmin_k ( ||c_k||^2 - 2 * x_n . c_k )
// bf16 WMMA GEMM, f32 accumulation. Each wave holds TWO A panels (32 rows,
// 256 VGPRs) so every B fragment pair feeds 4 WMMAs -> half the LDS traffic,
// 4 independent accumulator chains. Center tiles double-buffered in LDS via
// GLOBAL_LOAD_ASYNC_TO_LDS_B128 (ASYNCcnt), zero staging VGPRs.
// ---------------------------------------------------------------------------

typedef __attribute__((ext_vector_type(16))) __bf16 v16bf;
typedef __attribute__((ext_vector_type(8)))  float  v8f;

#define DIMS   512
#define ROWSU  260                 // LDS row stride in uints: 256 + 4 pad (stride%64==4 -> conflict-free b128)
#define TILEU  (32 * ROWSU)        // one 32-cluster tile in uints

static __device__ __forceinline__ unsigned pack_bf16x2(float a, float b) {
  unsigned ua = __float_as_uint(a), ub = __float_as_uint(b);
  ua = (ua + 0x7FFFu + ((ua >> 16) & 1u)) >> 16;   // round-to-nearest-even
  ub = (ub + 0x7FFFu + ((ub >> 16) & 1u)) >> 16;
  return ua | (ub << 16);
}

union FragU { unsigned w[8]; struct { uint4 lo, hi; } q; v16bf v; };

// LDS byte-offset of a generic pointer into shared memory
static __device__ __forceinline__ unsigned lds_off(const void* p) {
  return (unsigned)(unsigned long long)
         (__attribute__((address_space(3))) const void*)p;
}

// global -> LDS async copy, 16B per lane, tracked by ASYNCcnt
static __device__ __forceinline__ void async_g2l_b128(void* lds_dst, const void* gsrc) {
  unsigned            loff = lds_off(lds_dst);
  unsigned long long  ga   = (unsigned long long)
                             (__attribute__((address_space(1))) const void*)gsrc;
  asm volatile("global_load_async_to_lds_b128 %0, %1, off"
               :: "v"(loff), "v"(ga) : "memory");
}

static __device__ __forceinline__ void wait_async0() {
#if __has_builtin(__builtin_amdgcn_s_wait_asynccnt)
  __builtin_amdgcn_s_wait_asynccnt(0);
#else
  asm volatile("s_wait_asynccnt 0x0" ::: "memory");
#endif
}

// load one 16-row A panel (bf16 fragments) straight from global f32
static __device__ __forceinline__ void load_a_panel(
    const float* __restrict__ X, size_t row, int g, v16bf* afrag) {
  const float4* xr = (const float4*)(X + row * (size_t)DIMS);
#pragma unroll
  for (int s = 0; s < 16; ++s) {
    float4 a0 = xr[s * 8 + 2 * g + 0];       // K = s*32+8g   .. +3
    float4 a1 = xr[s * 8 + 2 * g + 1];       // K = s*32+8g+4 .. +7
    float4 b0 = xr[s * 8 + 4 + 2 * g + 0];   // K = s*32+16+8g ..
    float4 b1 = xr[s * 8 + 4 + 2 * g + 1];
    FragU f;
    f.w[0] = pack_bf16x2(a0.x, a0.y); f.w[1] = pack_bf16x2(a0.z, a0.w);
    f.w[2] = pack_bf16x2(a1.x, a1.y); f.w[3] = pack_bf16x2(a1.z, a1.w);
    f.w[4] = pack_bf16x2(b0.x, b0.y); f.w[5] = pack_bf16x2(b0.z, b0.w);
    f.w[6] = pack_bf16x2(b1.x, b1.y); f.w[7] = pack_bf16x2(b1.z, b1.w);
    afrag[s] = f.v;
  }
}

// --- Kernel 1: centers -> bf16 (workspace) + ||c||^2 (f32, exact) ----------
__global__ void __launch_bounds__(256) kmeans_prep_centers(
    const float* __restrict__ C, unsigned* __restrict__ cw,
    float* __restrict__ c2, int K) {
  const int wave = threadIdx.x >> 5;
  const int lane = threadIdx.x & 31;
  const int k = blockIdx.x * 8 + wave;
  if (k >= K) return;
  const float4* src = (const float4*)(C + (size_t)k * DIMS);
  unsigned*     dst = cw + (size_t)k * (DIMS / 2);
  float s = 0.0f;
#pragma unroll
  for (int j = 0; j < 4; ++j) {                 // 16 floats per lane
    float4 v = src[lane * 4 + j];
    s += v.x * v.x + v.y * v.y + v.z * v.z + v.w * v.w;
    dst[lane * 8 + j * 2 + 0] = pack_bf16x2(v.x, v.y);
    dst[lane * 8 + j * 2 + 1] = pack_bf16x2(v.z, v.w);
  }
#pragma unroll
  for (int off = 16; off >= 1; off >>= 1) s += __shfl_xor(s, off, 32);
  if (lane == 0) c2[k] = s;
}

// --- Kernel 2: assignment. 256 thr = 8 waves; block owns 256 rows of x. ----
__global__ void __launch_bounds__(256, 1) kmeans_assign(
    const float* __restrict__ X, const unsigned* __restrict__ cw,
    const float* __restrict__ c2, float* __restrict__ out, int K) {
  extern __shared__ unsigned cl[];       // 2 * TILEU uints: double-buffered center tile

  const int t    = threadIdx.x;
  const int lane = t & 31;
  const int wave = t >> 5;
  const int g    = lane >> 4;            // lane half (0/1)
  const int l16  = lane & 15;
  const size_t row0 = (size_t)blockIdx.x * 256;

  const uint4* cw4 = (const uint4*)cw;   // 64 uint4 per center row
  const int ntiles = K >> 5;             // 32 clusters per tile

  // prologue: async-stage tile 0 into buffer 0 (overlaps the A-panel loads)
#pragma unroll
  for (int i = 0; i < 8; ++i) {
    int idx = t + i * 256;
    async_g2l_b128(cl + (idx >> 6) * ROWSU + (idx & 63) * 4, cw4 + idx);
  }

  // ---- two A panels (32 rows) in 256 VGPRs -------------------------------
  v16bf afrag0[16], afrag1[16];
  load_a_panel(X, row0 + wave * 32 + l16,      g, afrag0);   // rows +0..15
  load_a_panel(X, row0 + wave * 32 + 16 + l16, g, afrag1);   // rows +16..31

  float bestv0[8], bestv1[8];
  int   besti0[8], besti1[8];
#pragma unroll
  for (int r = 0; r < 8; ++r) {
    bestv0[r] = 3.0e38f; besti0[r] = 0;
    bestv1[r] = 3.0e38f; besti1[r] = 0;
  }

  wait_async0();                          // own slice of tile 0 landed
  __syncthreads();                        // everyone's slice landed

  for (int ct = 0; ct < ntiles; ++ct) {
    unsigned* cur = cl + (ct & 1) * TILEU;
    unsigned* nxt = cl + ((ct + 1) & 1) * TILEU;
    const int c0 = ct << 5;
    const bool more = (ct + 1) < ntiles;

    // kick next tile's async global->LDS transfers; they ride under the WMMAs
    if (more) {
      const uint4* src = cw4 + (size_t)(c0 + 32) * 64;
#pragma unroll
      for (int i = 0; i < 8; ++i) {
        int idx = t + i * 256;
        async_g2l_b128(nxt + (idx >> 6) * ROWSU + (idx & 63) * 4, src + idx);
      }
      if (ct + 2 < ntiles)
        __builtin_prefetch(cw4 + (size_t)(c0 + 64) * 64 + t, 0, 0);
    }

    // ---- 4 x 16 WMMAs: each B fragment pair feeds 4 WMMAs ----
    v8f acc00 = {}, acc01 = {}, acc10 = {}, acc11 = {};
    const unsigned* b0p = cur + l16 * ROWSU;          // col = cluster c0+l16
    const unsigned* b1p = cur + (16 + l16) * ROWSU;   // col = cluster c0+16+l16
#pragma unroll
    for (int s = 0; s < 16; ++s) {
      // B 16-bit 32x16 layout: lane holds col N=l16, K = [s*32+8g, +8) and [s*32+16+8g, +8)
      FragU fb0, fb1;
      fb0.q.lo = *(const uint4*)(b0p + s * 16 + g * 4);
      fb0.q.hi = *(const uint4*)(b0p + s * 16 + g * 4 + 8);
      fb1.q.lo = *(const uint4*)(b1p + s * 16 + g * 4);
      fb1.q.hi = *(const uint4*)(b1p + s * 16 + g * 4 + 8);
      acc00 = __builtin_amdgcn_wmma_f32_16x16x32_bf16(false, afrag0[s], false, fb0.v,
                                                      (short)0, acc00, false, false);
      acc10 = __builtin_amdgcn_wmma_f32_16x16x32_bf16(false, afrag1[s], false, fb0.v,
                                                      (short)0, acc10, false, false);
      acc01 = __builtin_amdgcn_wmma_f32_16x16x32_bf16(false, afrag0[s], false, fb1.v,
                                                      (short)0, acc01, false, false);
      acc11 = __builtin_amdgcn_wmma_f32_16x16x32_bf16(false, afrag1[s], false, fb1.v,
                                                      (short)0, acc11, false, false);
    }

    const float c2a = c2[c0 + l16];
    const float c2b = c2[c0 + 16 + l16];
#pragma unroll
    for (int r = 0; r < 8; ++r) {        // D frag: lane -> (row blk+r+8g, col l16)
      float s00 = fmaf(-2.0f, acc00[r], c2a);
      float s01 = fmaf(-2.0f, acc01[r], c2b);
      float s10 = fmaf(-2.0f, acc10[r], c2a);
      float s11 = fmaf(-2.0f, acc11[r], c2b);
      if (s00 < bestv0[r]) { bestv0[r] = s00; besti0[r] = c0 + l16; }
      if (s01 < bestv0[r]) { bestv0[r] = s01; besti0[r] = c0 + 16 + l16; }
      if (s10 < bestv1[r]) { bestv1[r] = s10; besti1[r] = c0 + l16; }
      if (s11 < bestv1[r]) { bestv1[r] = s11; besti1[r] = c0 + 16 + l16; }
    }

    if (more) wait_async0();             // own async stores to nxt complete
    __syncthreads();                     // all waves' stores to nxt complete
  }

  // ---- reduce argmin across the 16 columns in each lane half ----
#pragma unroll
  for (int h = 0; h < 2; ++h) {
#pragma unroll
    for (int r = 0; r < 8; ++r) {
      float v = h ? bestv1[r] : bestv0[r];
      int   i = h ? besti1[r] : besti0[r];
#pragma unroll
      for (int off = 8; off >= 1; off >>= 1) {
        float ov = __shfl_xor(v, off, 32);
        int   oi = __shfl_xor(i, off, 32);
        if (ov < v || (ov == v && oi < i)) { v = ov; i = oi; }  // first-occurrence tie
      }
      if (l16 == 0)
        out[row0 + wave * 32 + h * 16 + g * 8 + r] = (float)i;
    }
  }
}

extern "C" void kernel_launch(void* const* d_in, const int* in_sizes, int n_in,
                              void* d_out, int out_size, void* d_ws, size_t ws_size,
                              hipStream_t stream) {
  const float* x = (const float*)d_in[0];
  const float* c = (const float*)d_in[1];
  const int N = in_sizes[0] / DIMS;      // 131072
  const int K = in_sizes[1] / DIMS;      // 2048

  // workspace: [bf16 centers: K*512*2 B][c2: K*4 B]
  unsigned* cw = (unsigned*)d_ws;
  float*    c2 = (float*)((char*)d_ws + (size_t)K * DIMS * 2);

  kmeans_prep_centers<<<(K + 7) / 8, 256, 0, stream>>>(c, cw, c2, K);

  const size_t shmem = (size_t)2 * TILEU * 4;   // 66,560 B  (<< 320 KB/WGP)
  kmeans_assign<<<N / 256, 256, shmem, stream>>>(x, cw, c2, (float*)d_out, K);
}
